// LSTMCell_28037546508956
// MI455X (gfx1250) — matrix operand order repaired and checked by reading, not compiled, not run on hardware
//
#include <hip/hip_runtime.h>

// ---------------------------------------------------------------------------
// LSTM cell, B=4096, IN=2048, H=2048 on MI455X (gfx1250, wave32, WMMA).
// One fused bf16 GEMM  gates[4096 x 8192] = A[4096 x 4096] @ W'^T  with
// gate-interleaved weight rows; LSTM nonlinearity fused into the epilogue.
// CDNA5 paths: v_wmma_f32_16x16x32_bf16 + global_load_async_to_lds_b128
// (ASYNCcnt-tracked 3-stage, K=64-deep LDS pipeline) + ds_load_b128 reads.
// ---------------------------------------------------------------------------

typedef __attribute__((ext_vector_type(16))) __bf16        v16bf;
typedef __attribute__((ext_vector_type(8)))  __bf16        v8bf;
typedef __attribute__((ext_vector_type(8)))  float         v8f;
typedef __attribute__((ext_vector_type(4)))  unsigned int  v4u;

#define B_DIM  4096
#define IN_DIM 2048
#define H_DIM  2048
#define K_TOT  (IN_DIM + H_DIM)   // 4096
#define N_TOT  (4 * H_DIM)        // 8192 (4 gates x H)

#define STAGE_K     64u                     // K-depth per pipeline stage
#define ROW_BYTES   144u                    // 128B data + 16B pad (36-bank stride, conflict free)
#define TILE_BYTES  (128u * ROW_BYTES)      // 18432 B per A or B tile
#define STAGE_BYTES (2u * TILE_BYTES)       // A + B per stage = 36864 B
#define NSTAGE      3                       // 108 KB of 320 KB WGP LDS
#define NSTEP       (K_TOT / STAGE_K)       // 64 stages

union Frag16 { v16bf v; v4u u[2]; };

// ---------------------------------------------------------------------------
// pack_A: A_bf16[b, k] = bf16( k < IN ? incoming[b,k] : h[b,k-IN] )
// ---------------------------------------------------------------------------
__global__ __launch_bounds__(256)
void pack_A(const float* __restrict__ x, const float* __restrict__ h,
            __bf16* __restrict__ A) {
    size_t idx  = (size_t)blockIdx.x * blockDim.x + threadIdx.x;
    size_t base = idx * 8;
    int b = (int)(base >> 12);
    int k = (int)(base & 4095);
    const float* src = (k < IN_DIM)
        ? (x + (size_t)b * IN_DIM + k)
        : (h + (size_t)b * H_DIM + (k - IN_DIM));
    float4 f0 = ((const float4*)src)[0];
    float4 f1 = ((const float4*)src)[1];
    v8bf o;
    o[0] = (__bf16)f0.x; o[1] = (__bf16)f0.y; o[2] = (__bf16)f0.z; o[3] = (__bf16)f0.w;
    o[4] = (__bf16)f1.x; o[5] = (__bf16)f1.y; o[6] = (__bf16)f1.z; o[7] = (__bf16)f1.w;
    *(v8bf*)(A + base) = o;
}

// ---------------------------------------------------------------------------
// pack_W: permuted bf16 weights. row n' = jgroup*64 + g*16 + jj
// (j = jgroup*16 + jj, g in {i,o,f,c}); col k: x-side then h-side.
// ---------------------------------------------------------------------------
struct WPtrs { const float* wx[4]; const float* wh[4]; };

__global__ __launch_bounds__(256)
void pack_W(WPtrs wp, __bf16* __restrict__ Wp) {
    size_t idx  = (size_t)blockIdx.x * blockDim.x + threadIdx.x;
    size_t base = idx * 8;
    int np = (int)(base >> 12);
    int k  = (int)(base & 4095);
    int jgroup = np >> 6;
    int rem    = np & 63;
    int g      = rem >> 4;
    int j      = (jgroup << 4) + (rem & 15);
    const float* src = (k < IN_DIM)
        ? (wp.wx[g] + (size_t)j * IN_DIM + k)
        : (wp.wh[g] + (size_t)j * H_DIM + (k - IN_DIM));
    float4 f0 = ((const float4*)src)[0];
    float4 f1 = ((const float4*)src)[1];
    v8bf o;
    o[0] = (__bf16)f0.x; o[1] = (__bf16)f0.y; o[2] = (__bf16)f0.z; o[3] = (__bf16)f0.w;
    o[4] = (__bf16)f1.x; o[5] = (__bf16)f1.y; o[6] = (__bf16)f1.z; o[7] = (__bf16)f1.w;
    *(v8bf*)(Wp + base) = o;
}

// ---------------------------------------------------------------------------
// CDNA5 async copy: 16 bytes global -> LDS, tracked by ASYNCcnt.
// ---------------------------------------------------------------------------
__device__ __forceinline__ void async_copy16(unsigned int lds_off,
                                             unsigned long long gaddr) {
    asm volatile("global_load_async_to_lds_b128 %0, %1, off"
                 :: "v"(lds_off), "v"(gaddr) : "memory");
}

__device__ __forceinline__ float fast_sigmoid(float x) {
    return 1.0f / (1.0f + __expf(-x));
}
__device__ __forceinline__ float fast_tanh(float x) {
    float e = __expf(2.0f * x);
    return (e - 1.0f) / (e + 1.0f);
}

// ---------------------------------------------------------------------------
// Fused GEMM + LSTM epilogue.
// Block: 256 threads = 8 waves, tile 128(M) x 128(N'); wave tile 32 x 64.
// 3-stage async LDS pipeline, each stage 64 K-deep; fills lead by 2 stages.
// ---------------------------------------------------------------------------
__global__ __launch_bounds__(256)
void lstm_gemm(const __bf16* __restrict__ A, const __bf16* __restrict__ W,
               const float* __restrict__ bi, const float* __restrict__ bo,
               const float* __restrict__ bfv_in, const float* __restrict__ bc,
               const float* __restrict__ c_old,
               float* __restrict__ h_out, float* __restrict__ c_out) {
    __shared__ __align__(16) unsigned char smem[NSTAGE * STAGE_BYTES]; // 108 KB

    const int tid  = threadIdx.x;
    const int lane = tid & 31;
    const int wave = tid >> 5;               // 0..7
    const int wm   = wave & 3;               // 4 waves along M
    const int wn   = wave >> 2;              // 2 waves along N'
    const int mBlock = blockIdx.y * 128;
    const int nBlock = blockIdx.x * 128;
    const int jj   = lane & 15;
    const int half = lane >> 4;

    // ---- async fill assignment: 8 x 16B per thread per stage ----
    // Tile = 128 rows x 128B. chunk = row*8 + c8; thread t -> rows t/8 + {0,32,64,96}.
    const int rowF = tid >> 3;               // 0..31
    const int c8F  = tid & 7;                // 16B chunk within row
    const __bf16* gA[4];
    const __bf16* gB[4];
    unsigned int lA[4], lB[4];
    const unsigned int ldsBase = (unsigned int)(uintptr_t)(void*)smem;
#pragma unroll
    for (int i = 0; i < 4; ++i) {
        const int r = rowF + 32 * i;
        gA[i] = A + (size_t)(mBlock + r) * K_TOT + c8F * 8;
        gB[i] = W + (size_t)(nBlock + r) * K_TOT + c8F * 8;
        lA[i] = ldsBase + (unsigned int)r * ROW_BYTES + c8F * 16u;
        lB[i] = lA[i] + TILE_BYTES;
    }

    // ---- per-lane fragment byte offsets inside a stage ----
    // A frag (16-bit 16x32): lane<16 holds K 0..7,16..23; lane>=16 K 8..15,24..31
    unsigned int aOff[2], bOff[4];
#pragma unroll
    for (int msub = 0; msub < 2; ++msub)
        aOff[msub] = (unsigned int)(wm * 32 + msub * 16 + jj) * ROW_BYTES + half * 16u;
#pragma unroll
    for (int t = 0; t < 4; ++t)
        bOff[t] = TILE_BYTES +
                  (unsigned int)(wn * 64 + t * 16 + jj) * ROW_BYTES + half * 16u;

    v8f acc[2][4] = {};   // [msub][gate]

    // ---- prologue: fill stages 0 and 1 ----
#pragma unroll
    for (int p = 0; p < 2; ++p) {
        const unsigned int s = (unsigned int)p * STAGE_BYTES;
        const int k0 = p * (int)STAGE_K;
#pragma unroll
        for (int i = 0; i < 4; ++i) {
            async_copy16(lA[i] + s, (unsigned long long)(uintptr_t)(gA[i] + k0));
            async_copy16(lB[i] + s, (unsigned long long)(uintptr_t)(gB[i] + k0));
        }
    }

    int buf = 0;
    for (int step = 0; step < NSTEP; ++step) {
        // wait until this step's stage is filled; next fill may stay in flight
        if (step + 1 < NSTEP) {
            asm volatile("s_wait_asynccnt 0x8" ::: "memory");
        } else {
            asm volatile("s_wait_asynccnt 0x0" ::: "memory");
        }
        __syncthreads();   // fills visible to all waves; stage (step-1) fully read

        // issue fill for step+2 (reuses the stage that was read at step-1)
        if (step + 2 < NSTEP) {
            const unsigned int s =
                (unsigned int)((step + 2) % NSTAGE) * STAGE_BYTES;
            const int k0 = (step + 2) * (int)STAGE_K;
#pragma unroll
            for (int i = 0; i < 4; ++i) {
                async_copy16(lA[i] + s, (unsigned long long)(uintptr_t)(gA[i] + k0));
                async_copy16(lB[i] + s, (unsigned long long)(uintptr_t)(gB[i] + k0));
            }
        }

        // ---- consume stage `buf`: two K=32 substeps, 16 WMMAs ----
        const unsigned int s = (unsigned int)buf * STAGE_BYTES;
#pragma unroll
        for (int ks = 0; ks < 2; ++ks) {
            const unsigned int kc = s + (unsigned int)ks * 64u;  // 32 bf16 = 64B
            Frag16 a0, a1, bfr[4];
            a0.u[0] = *(const v4u*)&smem[kc + aOff[0]];
            a0.u[1] = *(const v4u*)&smem[kc + aOff[0] + 32u];
            a1.u[0] = *(const v4u*)&smem[kc + aOff[1]];
            a1.u[1] = *(const v4u*)&smem[kc + aOff[1] + 32u];
#pragma unroll
            for (int t = 0; t < 4; ++t) {
                bfr[t].u[0] = *(const v4u*)&smem[kc + bOff[t]];
                bfr[t].u[1] = *(const v4u*)&smem[kc + bOff[t] + 32u];
            }
#pragma unroll
            for (int t = 0; t < 4; ++t) {
                acc[0][t] = __builtin_amdgcn_wmma_f32_16x16x32_bf16(
                    false, a0.v, false, bfr[t].v, (short)0, acc[0][t], false, false);
                acc[1][t] = __builtin_amdgcn_wmma_f32_16x16x32_bf16(
                    false, a1.v, false, bfr[t].v, (short)0, acc[1][t], false, false);
            }
        }
        buf = (buf == NSTAGE - 1) ? 0 : buf + 1;
    }

    // ---- fused LSTM epilogue ----
    // C/D layout: VGPR r, lanes0-15 -> M=r, lanes16-31 -> M=8+r; N = lane&15.
    const int j = ((nBlock + wn * 64) >> 6) * 16 + jj;   // hidden-unit index
    const float biv = bi[j];
    const float bov = bo[j];
    const float bfv = bfv_in[j];
    const float bcv = bc[j];

#pragma unroll
    for (int msub = 0; msub < 2; ++msub) {
        const int mrow0 = mBlock + wm * 32 + msub * 16 + half * 8;
#pragma unroll
        for (int r = 0; r < 8; ++r) {
            const int m = mrow0 + r;
            const size_t off = (size_t)m * H_DIM + j;
            float i_t = fast_sigmoid(acc[msub][0][r] + biv);
            float o_t = fast_sigmoid(acc[msub][1][r] + bov);
            float f_t = fast_sigmoid(acc[msub][2][r] + bfv);
            float cg  = fast_tanh   (acc[msub][3][r] + bcv);
            float cn  = f_t * c_old[off] + i_t * cg;
            float hn  = o_t * fast_tanh(cn);
            h_out[off] = hn;
            c_out[off] = cn;
        }
    }
}

// ---------------------------------------------------------------------------
extern "C" void kernel_launch(void* const* d_in, const int* in_sizes, int n_in,
                              void* d_out, int out_size, void* d_ws, size_t ws_size,
                              hipStream_t stream) {
    const float* incoming = (const float*)d_in[0];
    const float* h        = (const float*)d_in[1];
    const float* c        = (const float*)d_in[2];
    const float* Wxi = (const float*)d_in[3];
    const float* Wxo = (const float*)d_in[4];
    const float* Wxf = (const float*)d_in[5];
    const float* Wxc = (const float*)d_in[6];
    const float* bi  = (const float*)d_in[7];
    const float* bo  = (const float*)d_in[8];
    const float* bf  = (const float*)d_in[9];
    const float* bc  = (const float*)d_in[10];
    const float* Whi = (const float*)d_in[11];
    const float* Who = (const float*)d_in[12];
    const float* Whf = (const float*)d_in[13];
    const float* Whc = (const float*)d_in[14];

    __bf16* Abf = (__bf16*)d_ws;                                      // 32 MB
    __bf16* Wbf = (__bf16*)((char*)d_ws + (size_t)B_DIM * K_TOT * 2); // 64 MB

    float* h_out = (float*)d_out;
    float* c_out = h_out + (size_t)B_DIM * H_DIM;

    // 1) pack activations: [x | h] -> bf16 A[4096 x 4096]
    {
        int threads = (B_DIM * K_TOT) / 8;
        pack_A<<<threads / 256, 256, 0, stream>>>(incoming, h, Abf);
    }
    // 2) pack gate-interleaved bf16 weights W'[8192 x 4096]
    {
        WPtrs wp;
        wp.wx[0] = Wxi; wp.wx[1] = Wxo; wp.wx[2] = Wxf; wp.wx[3] = Wxc;
        wp.wh[0] = Whi; wp.wh[1] = Who; wp.wh[2] = Whf; wp.wh[3] = Whc;
        int threads = (N_TOT * K_TOT) / 8;
        pack_W<<<threads / 256, 256, 0, stream>>>(wp, Wbf);
    }
    // 3) fused WMMA GEMM + LSTM epilogue
    {
        dim3 grid(N_TOT / 128, B_DIM / 128);   // (64, 32)
        lstm_gemm<<<grid, 256, 0, stream>>>(Abf, Wbf, bi, bo, bf, bc,
                                            c, h_out, c_out);
    }
}
